// DSAIndexer_25744033972324
// MI455X (gfx1250) — compile-verified, matrix operand order checked by reading, not run on hardware
//
#include <hip/hip_runtime.h>
#include <hip/hip_bf16.h>
#include <math.h>

typedef __attribute__((ext_vector_type(16))) _Float16 v16h;
typedef __attribute__((ext_vector_type(8)))  _Float16 v8h;
typedef __attribute__((ext_vector_type(4)))  _Float16 v4h;
typedef __attribute__((ext_vector_type(8)))  float    v8f;
typedef __attribute__((ext_vector_type(4)))  float    v4f;
typedef __attribute__((ext_vector_type(4)))  unsigned int u32x4;
typedef __attribute__((ext_vector_type(8)))  int      i32x8;
typedef __attribute__((ext_vector_type(4)))  int      i32x4;

#define SEQ 4096
#define DM  2048
#define DCQ 1536
#define HH  8
#define DD  128
#define QROW (HH * DD)               // 1024 halves per t-row
#define NEGV (-1000000000.0f)
#define INV_SQRT_D 0.08838834764831845f   // 128^-0.5 (also Hadamard scale)

#if defined(__has_builtin)
#if __has_builtin(__builtin_amdgcn_tensor_load_to_lds) && __has_builtin(__builtin_amdgcn_s_wait_tensorcnt)
#define HAVE_TDM 1
#endif
#endif
#ifndef HAVE_TDM
#define HAVE_TDM 0
#endif

// ---- fragment loader: 8 contiguous halves at p, 8 more at p+16 (ISA 16-bit A/B layout) ----
__device__ __forceinline__ v16h load_frag16(const _Float16* p) {
  v8h lo = *(const v8h*)(p);
  v8h hi = *(const v8h*)(p + 16);
  v16h r;
#pragma unroll
  for (int i = 0; i < 8; ++i) { r[i] = lo[i]; r[i + 8] = hi[i]; }
  return r;
}

// ---- f32 -> f16 convert ----
__global__ void cvt_f32_f16(const float* __restrict__ in, _Float16* __restrict__ out, int n) {
  int i = blockIdx.x * blockDim.x + threadIdx.x;
  if (i < n) out[i] = (_Float16)in[i];
}

// ---- f32 [K][N] -> f16 [N][K] transpose-convert ----
__global__ void tcvt_f32_f16(const float* __restrict__ in, _Float16* __restrict__ out, int K, int N) {
  int i = blockIdx.x * blockDim.x + threadIdx.x;
  if (i >= K * N) return;
  int n = i / K, k = i - n * K;
  out[i] = (_Float16)in[(size_t)k * N + n];
}

// ---- WMMA GEMM: C[M][N] = A[M][K](f16,row-major) x Bt[N][K](f16) ; wave does 16x64 ----
__global__ void gemm_f16(const _Float16* __restrict__ A, const _Float16* __restrict__ Bt,
                         float* __restrict__ C, int M, int N, int K) {
  const int lane = threadIdx.x & 31;
  const int wave = threadIdx.x >> 5;
  const int l15  = lane & 15;
  const int hio  = (lane >> 4) * 8;
  const int m0 = blockIdx.y * 128 + wave * 16;
  const int n0 = blockIdx.x * 64;
  v8f acc[4] = {};
  const _Float16* arow = A  + (size_t)(m0 + l15) * K + hio;
  const _Float16* brow = Bt + (size_t)(n0 + l15) * K + hio;
  for (int kc = 0; kc < K; kc += 32) {
    v16h a = load_frag16(arow + kc);
#pragma unroll
    for (int j = 0; j < 4; ++j) {
      v16h b = load_frag16(brow + (size_t)(16 * j) * K + kc);
      acc[j] = __builtin_amdgcn_wmma_f32_16x16x32_f16(false, a, false, b, (short)0, acc[j], false, false);
    }
  }
  const int rbase = m0 + hio;   // C layout: VGPR i -> M = i (+8 for lanes 16-31)
#pragma unroll
  for (int i = 0; i < 8; ++i) {
    float* cr = C + (size_t)(rbase + i) * N + n0 + l15;
#pragma unroll
    for (int j = 0; j < 4; ++j) cr[16 * j] = acc[j][i];
  }
}

// ---- RoPE(first 64 dims) + 128-pt Hadamard, wave-per-row, 4 elems/lane (d = lane*4+j) ----
__device__ __forceinline__ v4f rope_had(v4f v, int lane, float pos) {
  float pr[4];
#pragma unroll
  for (int j = 0; j < 4; ++j) pr[j] = __shfl_xor(v[j], 8, 32);
  if (lane < 16) {
    const int base = (lane & 7) * 4;
    const float sgn = (lane < 8) ? -1.0f : 1.0f;
#pragma unroll
    for (int j = 0; j < 4; ++j) {
      float fr = powf(10000.0f, -(float)(base + j) * (1.0f / 32.0f));
      float sn, cs;
      sincosf(pos * fr, &sn, &cs);
      v[j] = v[j] * cs + sgn * pr[j] * sn;
    }
  }
  { float a0 = v[0], a1 = v[1], a2 = v[2], a3 = v[3];
    v[0] = a0 + a1; v[1] = a0 - a1; v[2] = a2 + a3; v[3] = a2 - a3; }
  { float a0 = v[0], a1 = v[1], a2 = v[2], a3 = v[3];
    v[0] = a0 + a2; v[1] = a1 + a3; v[2] = a0 - a2; v[3] = a1 - a3; }
#pragma unroll
  for (int m = 1; m <= 16; m <<= 1) {
#pragma unroll
    for (int j = 0; j < 4; ++j) {
      float p = __shfl_xor(v[j], m, 32);
      v[j] = (lane & m) ? (p - v[j]) : (v[j] + p);
    }
  }
#pragma unroll
  for (int j = 0; j < 4; ++j) v[j] *= INV_SQRT_D;
  return v;
}

// ---- k path: LayerNorm + RoPE + Hadamard -> f16 ----
__global__ void kprep(const float* __restrict__ kraw, const float* __restrict__ g,
                      const float* __restrict__ bt, _Float16* __restrict__ kf16) {
  const int lane = threadIdx.x & 31;
  const int s = blockIdx.x * 8 + (threadIdx.x >> 5);
  v4f v = *(const v4f*)(kraw + (size_t)s * DD + lane * 4);
  float sum = v[0] + v[1] + v[2] + v[3];
#pragma unroll
  for (int o = 16; o >= 1; o >>= 1) sum += __shfl_xor(sum, o, 32);
  float mu = sum * (1.0f / 128.0f);
  float ls = 0.f;
#pragma unroll
  for (int j = 0; j < 4; ++j) { float d = v[j] - mu; ls += d * d; }
#pragma unroll
  for (int o = 16; o >= 1; o >>= 1) ls += __shfl_xor(ls, o, 32);
  float rstd = rsqrtf(ls * (1.0f / 128.0f) + 1e-5f);
  v4f gg = *(const v4f*)(g + lane * 4);
  v4f bb = *(const v4f*)(bt + lane * 4);
#pragma unroll
  for (int j = 0; j < 4; ++j) v[j] = (v[j] - mu) * rstd * gg[j] + bb[j];
  v = rope_had(v, lane, (float)s);
  v4h o;
#pragma unroll
  for (int j = 0; j < 4; ++j) o[j] = (_Float16)v[j];
  *(v4h*)(kf16 + (size_t)s * DD + lane * 4) = o;
}

// ---- q path: RoPE + Hadamard -> f16 (row = t*8 + h) ----
__global__ void qprep(const float* __restrict__ qf32, _Float16* __restrict__ qf16) {
  const int lane = threadIdx.x & 31;
  const int row = blockIdx.x * 8 + (threadIdx.x >> 5);
  const int t = row >> 3;
  v4f v = *(const v4f*)(qf32 + (size_t)row * DD + lane * 4);
  v = rope_had(v, lane, (float)t);
  v4h o;
#pragma unroll
  for (int j = 0; j < 4; ++j) o[j] = (_Float16)v[j];
  *(v4h*)(qf16 + (size_t)row * DD + lane * 4) = o;
}

// ---- w[t][h] = x[t,:] . Wwt[:,h], wave-per-(t,h) ----
__global__ void wproj(const float* __restrict__ x, const float* __restrict__ Wwt,
                      float* __restrict__ w) {
  const int lane = threadIdx.x & 31;
  const int h = threadIdx.x >> 5;
  const int t = blockIdx.x;
  const float* xr = x + (size_t)t * DM;
  float s = 0.f;
  for (int d = lane; d < DM; d += 32) s += xr[d] * Wwt[d * HH + h];
#pragma unroll
  for (int o = 16; o >= 1; o >>= 1) s += __shfl_xor(s, o, 32);
  if (lane == 0) w[t * HH + h] = s;
}

// ---- fused attention: block = 8 waves sharing one t-tile (TDM-staged in LDS),
// ----                  each wave computes one 16x16 (t,s) tile across all 8 heads ----
__global__ void attn(const _Float16* __restrict__ qf16, const _Float16* __restrict__ kf16,
                     const float* __restrict__ w, float* __restrict__ out) {
  __shared__ _Float16 qs[16 * QROW];          // 32 KB: 16 t-rows x (8 heads * 128)
  const int lane = threadIdx.x & 31;
  const int wave = threadIdx.x >> 5;
  const int l15  = lane & 15;
  const int hio  = (lane >> 4) * 8;
  const int t0 = blockIdx.y * 16;
  const int s0 = (blockIdx.x * 8 + wave) * 16;
  const bool block_masked = (blockIdx.x * 128) > (t0 + 15);

  if (!block_masked) {
#if HAVE_TDM
    if (wave == 0) {
      // Tensor DMA: contiguous 2-D tile, 1024 f16 per row x 16 rows, stride 1024.
      unsigned long long ga = (unsigned long long)(uintptr_t)(qf16 + (size_t)t0 * QROW);
      unsigned int lds_addr = (unsigned int)(uintptr_t)(&qs[0]);   // low 32 bits = LDS offset
      u32x4 g0;
      g0[0] = 1u;                                            // count=1 (valid user D#)
      g0[1] = lds_addr;                                      // lds_addr (bytes)
      g0[2] = (unsigned int)ga;                              // global_addr[31:0]
      g0[3] = (unsigned int)((ga >> 32) & 0x1FFFFFFull) | (2u << 30);  // addr[56:32] | type=2
      i32x8 g1;
      g1[0] = (int)(1u << 16);          // workgroup_mask=0, data_size=1 (2 bytes)
      g1[1] = (int)(1024u << 16);       // tensor_dim0 = 1024 (bits 79:48, low half)
      g1[2] = (int)(4096u << 16);       // tensor_dim1 = 4096 (bits 111:80, low half)
      g1[3] = (int)(1024u << 16);       // tile_dim0 = 1024 (bits 127:112)
      g1[4] = 16;                       // tile_dim1 = 16 rows, tile_dim2 = 0
      g1[5] = 1024;                     // tensor_dim0_stride = 1024 (elements)
      g1[6] = 0;                        // stride high bits / dim1_stride low
      g1[7] = 0;
      i32x4 gz; gz[0] = 0; gz[1] = 0; gz[2] = 0; gz[3] = 0;
#if __clang_major__ >= 23
      i32x8 gz8;
#pragma unroll
      for (int i = 0; i < 8; ++i) gz8[i] = 0;
      __builtin_amdgcn_tensor_load_to_lds(g0, g1, gz, gz, gz8, 0);
#else
      __builtin_amdgcn_tensor_load_to_lds(g0, g1, gz, gz, 0);
#endif
      __builtin_amdgcn_s_wait_tensorcnt(0);
    }
#else
    { // fallback: cooperative 128-bit copy (16*1024 halves = 2048 x float4)
      const float4* src = (const float4*)(qf16 + (size_t)t0 * QROW);
      float4* dst = (float4*)qs;
      for (int i = threadIdx.x; i < 2048; i += 256) dst[i] = src[i];
    }
#endif
    __syncthreads();
  }

  if (s0 > t0 + 15) {                   // this wave's tile fully above diagonal
#pragma unroll
    for (int i = 0; i < 8; ++i)
      out[(size_t)(t0 + hio + i) * SEQ + s0 + l15] = NEGV;
    return;
  }

  v8f acc[8] = {};
  const _Float16* kb   = kf16 + (size_t)(s0 + l15) * DD + hio;   // B: per-wave k tile (global/L2)
  const _Float16* qrow = qs + (size_t)l15 * QROW + hio;          // A: shared q tile (LDS)
#pragma unroll
  for (int kc = 0; kc < DD; kc += 32) {
    v16h b = load_frag16(kb + kc);
#pragma unroll
    for (int h = 0; h < HH; ++h) {
      v16h a = load_frag16(qrow + h * DD + kc);
      acc[h] = __builtin_amdgcn_wmma_f32_16x16x32_f16(false, a, false, b, (short)0, acc[h], false, false);
    }
  }
#pragma unroll
  for (int i = 0; i < 8; ++i) {
    const int row = t0 + hio + i;
    v4f w0 = *(const v4f*)(w + row * HH);
    v4f w1 = *(const v4f*)(w + row * HH + 4);
    float s = 0.f;
#pragma unroll
    for (int h = 0; h < 4; ++h) s += w0[h] * fmaxf(acc[h][i], 0.f);
#pragma unroll
    for (int h = 0; h < 4; ++h) s += w1[h] * fmaxf(acc[h + 4][i], 0.f);
    const int col = s0 + l15;
    out[(size_t)row * SEQ + col] = (col <= row) ? s * INV_SQRT_D : NEGV;
  }
}

extern "C" void kernel_launch(void* const* d_in, const int* in_sizes, int n_in,
                              void* d_out, int out_size, void* d_ws, size_t ws_size,
                              hipStream_t stream) {
  (void)in_sizes; (void)n_in; (void)out_size; (void)ws_size;
  const float* x   = (const float*)d_in[0];
  const float* ql  = (const float*)d_in[1];
  const float* Wq  = (const float*)d_in[2];
  const float* Wk  = (const float*)d_in[3];
  const float* lng = (const float*)d_in[4];
  const float* lnb = (const float*)d_in[5];
  const float* Wwt = (const float*)d_in[6];
  float* out = (float*)d_out;

  char* ws = (char*)d_ws;
  size_t off = 0;
  auto take = [&](size_t bytes) -> char* {
    char* p = ws + off; off += (bytes + 255) & ~(size_t)255; return p;
  };
  // big region: x_f16 (phase 1) then q_f32 (phase 2) — both 16,777,216 bytes
  char* big = take((size_t)SEQ * DM * 2);
  _Float16* x16  = (_Float16*)big;
  float*    qf32 = (float*)big;
  // region: q_latent_f16 (phase 2 GEMM input) then q_f16 (after GEMM done)
  char* r2 = take((size_t)SEQ * DCQ * 2);
  _Float16* qlat16 = (_Float16*)r2;
  _Float16* qf16   = (_Float16*)r2;
  _Float16* Wqt16 = (_Float16*)take((size_t)DCQ * QROW * 2);
  _Float16* Wkt16 = (_Float16*)take((size_t)DM * DD * 2);
  float*    kraw  = (float*)take((size_t)SEQ * DD * 4);
  _Float16* kf16  = (_Float16*)take((size_t)SEQ * DD * 2);
  float*    wbuf  = (float*)take((size_t)SEQ * HH * 4);

  // ---- phase 1: k path ----
  cvt_f32_f16<<<(SEQ * DM + 255) / 256, 256, 0, stream>>>(x, x16, SEQ * DM);
  tcvt_f32_f16<<<(DM * DD + 255) / 256, 256, 0, stream>>>(Wk, Wkt16, DM, DD);
  gemm_f16<<<dim3(DD / 64, SEQ / 128), 256, 0, stream>>>(x16, Wkt16, kraw, SEQ, DD, DM);
  kprep<<<SEQ / 8, 256, 0, stream>>>(kraw, lng, lnb, kf16);
  // ---- phase 2: q path (x16 dead; big region becomes q_f32) ----
  cvt_f32_f16<<<(SEQ * DCQ + 255) / 256, 256, 0, stream>>>(ql, qlat16, SEQ * DCQ);
  tcvt_f32_f16<<<(DCQ * QROW + 255) / 256, 256, 0, stream>>>(Wq, Wqt16, DCQ, QROW);
  gemm_f16<<<dim3(QROW / 64, SEQ / 128), 256, 0, stream>>>(qlat16, Wqt16, qf32, SEQ, QROW, DCQ);
  qprep<<<(SEQ * HH) / 8, 256, 0, stream>>>(qf32, qf16);
  // ---- w projection (reads original f32 x) ----
  wproj<<<SEQ, 256, 0, stream>>>(x, Wwt, wbuf);
  // ---- fused causal attention scores (TDM-staged q tile) ----
  attn<<<dim3(SEQ / 16 / 8, SEQ / 16), 256, 0, stream>>>(qf16, kf16, wbuf, out);
}
